// edge_update_70961449664973
// MI455X (gfx1250) — compile-verified, matrix operand order checked by reading, not run on hardware
//
#include <hip/hip_runtime.h>

// ---------------------------------------------------------------------------
// Edge-update (SchNet-style CFConv edge weighting) for MI455X / gfx1250.
// GEMMs run on v_wmma_f32_16x16x32_bf16 (f32 accumulate); everything
// element-wise stays fp32.  Memory-bound target: ~470 MB HBM -> ~20 us.
// ---------------------------------------------------------------------------

typedef __attribute__((ext_vector_type(16))) __bf16 bf16x16;
typedef __attribute__((ext_vector_type(8)))  float  v8f;

#define E_EDGES 640000
#define NNODES  20000
#define HIDDEN  128
#define FILTERS 128
#define GAUSS   50
#define K1PAD   64            // GAUSS padded to a multiple of 32 for WMMA

union AFrag { bf16x16 v; float4 q[2]; };

// Load one 16-bit WMMA operand fragment (A-layout / mirrored B-layout) from an
// LDS row of bf16 values.  Per ISA 16-bit A 16x32 table: lane group g=lane>>4,
// element e -> K = (e%8) + 8*g + 16*(e/8)  => two contiguous 16B chunks.
__device__ __forceinline__ bf16x16 load_frag(const __bf16* row, int k0, int g) {
    AFrag f;
    f.q[0] = *reinterpret_cast<const float4*>(row + k0 + 8 * g);
    f.q[1] = *reinterpret_cast<const float4*>(row + k0 + 8 * g + 16);
    return f.v;
}

__device__ __forceinline__ v8f wmma_bf16(bf16x16 a, bf16x16 b, v8f c) {
    return __builtin_amdgcn_wmma_f32_16x16x32_bf16(false, a, false, b,
                                                   (short)0, c, false, false);
}

// ---------------------------------------------------------------------------
// Kernel 1: nf = node_feats @ W_lin^T   [20000,128] x [128,128]
// ---------------------------------------------------------------------------
__global__ __launch_bounds__(256) void nf_kernel(const float* __restrict__ node_feats,
                                                 const float* __restrict__ W_lin,
                                                 float* __restrict__ nf) {
    __shared__ __bf16 shB[FILTERS * HIDDEN];        // row n = W_lin[n][:]  (B[k][n])
    __shared__ __bf16 shA[8 * 16 * HIDDEN];         // per-wave node tiles
    __shared__ float  shO[8 * 16 * FILTERS];        // per-wave output stage

    const int tid  = threadIdx.x;
    const int lane = tid & 31;
    const int wave = tid >> 5;
    const int nl   = lane & 15;
    const int g    = lane >> 4;

    for (int i = tid * 4; i < FILTERS * HIDDEN; i += 256 * 4) {
        float4 w = *reinterpret_cast<const float4*>(W_lin + i);
        shB[i + 0] = (__bf16)w.x; shB[i + 1] = (__bf16)w.y;
        shB[i + 2] = (__bf16)w.z; shB[i + 3] = (__bf16)w.w;
    }

    const int row0 = blockIdx.x * 128 + wave * 16;
    __bf16* myA = shA + wave * 16 * HIDDEN;
    float*  myO = shO + wave * 16 * FILTERS;

    for (int rr = 0; rr < 16; ++rr) {
        int row = row0 + rr;
        float4 v = make_float4(0.f, 0.f, 0.f, 0.f);
        if (row < NNODES)
            v = *reinterpret_cast<const float4*>(node_feats + (long long)row * HIDDEN + lane * 4);
        __bf16* dst = myA + rr * HIDDEN + lane * 4;
        dst[0] = (__bf16)v.x; dst[1] = (__bf16)v.y;
        dst[2] = (__bf16)v.z; dst[3] = (__bf16)v.w;
    }
    __syncthreads();

    v8f acc[8];
#pragma unroll
    for (int t = 0; t < 8; ++t) acc[t] = (v8f)0.0f;

#pragma unroll
    for (int s = 0; s < 4; ++s) {
        const int k0 = s * 32;
        bf16x16 fa = load_frag(myA + nl * HIDDEN, k0, g);
#pragma unroll
        for (int t = 0; t < 8; ++t) {
            bf16x16 fb = load_frag(shB + (t * 16 + nl) * HIDDEN, k0, g);
            acc[t] = wmma_bf16(fa, fb, acc[t]);
        }
    }

#pragma unroll
    for (int t = 0; t < 8; ++t)
#pragma unroll
        for (int r = 0; r < 8; ++r)
            myO[(r + 8 * g) * FILTERS + t * 16 + nl] = acc[t][r];
    __syncthreads();

    for (int rr = 0; rr < 16; ++rr) {
        int row = row0 + rr;
        if (row < NNODES) {
            float4 v = *reinterpret_cast<const float4*>(myO + rr * FILTERS + lane * 4);
            *reinterpret_cast<float4*>(nf + (long long)row * FILTERS + lane * 4) = v;
        }
    }
}

// ---------------------------------------------------------------------------
// Kernel 2: per 128-edge block: GEMM1 -> softplus -> GEMM2 -> *cutoff ->
//           gather nf[j] and multiply -> coalesced store.
// ---------------------------------------------------------------------------
__global__ __launch_bounds__(256) void edge_kernel(const float* __restrict__ dist,
                                                   const float* __restrict__ rbf,
                                                   const long long* __restrict__ edge_index,
                                                   const float* __restrict__ W1,
                                                   const float* __restrict__ b1,
                                                   const float* __restrict__ W2,
                                                   const float* __restrict__ b2,
                                                   const float* __restrict__ nf,
                                                   float* __restrict__ out) {
    __shared__ __bf16 shW1[FILTERS * K1PAD];        // 16 KB, row n = padded W1[n][:]
    __shared__ __bf16 shW2[FILTERS * FILTERS];      // 32 KB, row n = W2[n][:]
    __shared__ __bf16 shA [8 * 16 * K1PAD];         // 16 KB, per-wave rbf tiles
    __shared__ __bf16 shH [8 * 16 * FILTERS];       // 32 KB, per-wave hidden acts
    __shared__ float  shO [8 * 16 * FILTERS];       // 64 KB, per-wave output stage
    __shared__ float  shC [8 * 16];                 // per-edge cutoff

    const int tid  = threadIdx.x;
    const int lane = tid & 31;
    const int wave = tid >> 5;
    const int nl   = lane & 15;
    const int g    = lane >> 4;
    const int eb   = blockIdx.x * 128 + wave * 16;  // wave's first edge (always in range)

    // Stage W1 (pad K 50 -> 64 with zeros)
    for (int i = tid; i < FILTERS * K1PAD; i += 256) {
        int n = i >> 6, k = i & 63;
        shW1[i] = (k < GAUSS) ? (__bf16)W1[n * GAUSS + k] : (__bf16)0.0f;
    }
    // Stage W2
    for (int i = tid * 4; i < FILTERS * FILTERS; i += 256 * 4) {
        float4 w = *reinterpret_cast<const float4*>(W2 + i);
        shW2[i + 0] = (__bf16)w.x; shW2[i + 1] = (__bf16)w.y;
        shW2[i + 2] = (__bf16)w.z; shW2[i + 3] = (__bf16)w.w;
    }
    // Stage rbf tile for this wave (16 edges x 64 padded K)
    __bf16* myA = shA + wave * 16 * K1PAD;
    for (int rr = 0; rr < 16; ++rr) {
        long long e = eb + rr;
#pragma unroll
        for (int kk = lane; kk < K1PAD; kk += 32) {
            float x = (kk < GAUSS) ? rbf[e * GAUSS + kk] : 0.0f;
            myA[rr * K1PAD + kk] = (__bf16)x;
        }
    }
    // Cosine cutoff for the wave's 16 edges
    if (lane < 16) {
        float d = dist[eb + lane];
        shC[wave * 16 + lane] = 0.5f * (cosf(d * 0.31415926535897932f) + 1.0f);
    }
    __syncthreads();

    // ---- GEMM1: [16 x 64] @ [64 x 128]  (rbf @ W1^T) ----
    v8f acc[8];
#pragma unroll
    for (int t = 0; t < 8; ++t) acc[t] = (v8f)0.0f;
#pragma unroll
    for (int s = 0; s < 2; ++s) {
        const int k0 = s * 32;
        bf16x16 fa = load_frag(myA + nl * K1PAD, k0, g);
#pragma unroll
        for (int t = 0; t < 8; ++t) {
            bf16x16 fb = load_frag(shW1 + (t * 16 + nl) * K1PAD, k0, g);
            acc[t] = wmma_bf16(fa, fb, acc[t]);
        }
    }

    // ---- bias + shifted softplus (fp32), store bf16 h-tile ----
    __bf16* myH = shH + wave * 16 * FILTERS;
#pragma unroll
    for (int t = 0; t < 8; ++t) {
        float bb = b1[t * 16 + nl];
#pragma unroll
        for (int r = 0; r < 8; ++r) {
            float x = acc[t][r] + bb;
            float v = fmaxf(x, 0.0f) + log1pf(__expf(-fabsf(x))) - 0.69314718055994531f;
            myH[(r + 8 * g) * FILTERS + t * 16 + nl] = (__bf16)v;
        }
    }
    __syncthreads();

    // ---- GEMM2: [16 x 128] @ [128 x 128]  (h @ W2^T) ----
#pragma unroll
    for (int t = 0; t < 8; ++t) acc[t] = (v8f)0.0f;
#pragma unroll
    for (int s = 0; s < 4; ++s) {
        const int k0 = s * 32;
        bf16x16 fa = load_frag(myH + nl * FILTERS, k0, g);
#pragma unroll
        for (int t = 0; t < 8; ++t) {
            bf16x16 fb = load_frag(shW2 + (t * 16 + nl) * FILTERS, k0, g);
            acc[t] = wmma_bf16(fa, fb, acc[t]);
        }
    }

    // ---- bias + cutoff scale, stage fp32 result ----
    float* myO = shO + wave * 16 * FILTERS;
#pragma unroll
    for (int r = 0; r < 8; ++r) {
        float Cm = shC[wave * 16 + r + 8 * g];
#pragma unroll
        for (int t = 0; t < 8; ++t) {
            float w = (acc[t][r] + b2[t * 16 + nl]) * Cm;
            myO[(r + 8 * g) * FILTERS + t * 16 + nl] = w;
        }
    }
    __syncthreads();

    // ---- gather nf[j], multiply, coalesced float4 store ----
    for (int rr = 0; rr < 16; ++rr) {
        long long e = eb + rr;
        long long j = edge_index[e];             // row 0 of [2, E]
        float4 v  = *reinterpret_cast<const float4*>(myO + rr * FILTERS + lane * 4);
        float4 nv = *reinterpret_cast<const float4*>(nf + j * FILTERS + lane * 4);
        float4 o;
        o.x = v.x * nv.x; o.y = v.y * nv.y; o.z = v.z * nv.z; o.w = v.w * nv.w;
        *reinterpret_cast<float4*>(out + e * FILTERS + lane * 4) = o;
    }
}

// ---------------------------------------------------------------------------
extern "C" void kernel_launch(void* const* d_in, const int* in_sizes, int n_in,
                              void* d_out, int out_size, void* d_ws, size_t ws_size,
                              hipStream_t stream) {
    (void)in_sizes; (void)n_in; (void)out_size; (void)ws_size;
    const float*     node_feats = (const float*)d_in[0];
    const float*     dist       = (const float*)d_in[1];
    const float*     rbf        = (const float*)d_in[2];
    const long long* edge_index = (const long long*)d_in[3];   // int64 per reference
    const float*     W_lin      = (const float*)d_in[4];
    const float*     W1         = (const float*)d_in[5];
    const float*     b1         = (const float*)d_in[6];
    const float*     W2         = (const float*)d_in[7];
    const float*     b2         = (const float*)d_in[8];
    float* out = (float*)d_out;
    float* nf  = (float*)d_ws;                 // 20000*128 f32 = 10.24 MB scratch

    nf_kernel<<<(NNODES + 127) / 128, 256, 0, stream>>>(node_feats, W_lin, nf);
    edge_kernel<<<E_EDGES / 128, 256, 0, stream>>>(dist, rbf, edge_index,
                                                   W1, b1, W2, b2, nf, out);
}
